// SlotAttention_66468913873547
// MI455X (gfx1250) — compile-verified, matrix operand order checked by reading, not run on hardware
//
#include <hip/hip_runtime.h>
#include <hip/hip_bf16.h>

// ---------------------------------------------------------------------------
// Types for CDNA5 WMMA (wave32): V_WMMA_F32_16X16X32_BF16
// ---------------------------------------------------------------------------
typedef __attribute__((ext_vector_type(16))) __bf16 v16bf;
typedef __attribute__((ext_vector_type(8)))  __bf16 v8bf;
typedef __attribute__((ext_vector_type(8)))  float  v8f;

#define DEV static __device__ __forceinline__

// Problem constants
#define BB   32
#define SS   11
#define SP   16      // padded slots per batch
#define FF   4096
#define DD   256
#define HID  1024
#define EPSA 1e-8f
#define LNEPS 1e-5f

DEV float wave_sum(float v) {
#pragma unroll
  for (int o = 16; o > 0; o >>= 1) v += __shfl_xor(v, o, 32);
  return v;
}

DEV float block_sum(float v, float* red) {
  int d = threadIdx.x;
  red[d] = v; __syncthreads();
  for (int o = 128; o > 0; o >>= 1) {
    if (d < o) red[d] += red[d + o];
    __syncthreads();
  }
  float r = red[0]; __syncthreads();
  return r;
}

DEV float sigmoidf(float x) { return 1.0f / (1.0f + __expf(-x)); }

// ---------------------------------------------------------------------------
// Fragment loader for C = A @ S^T, both A and S row-major [rows][K].
// Wave32 16-bit operand layout (ISA 7.12.2):
//   lanes 0-15 : row = lane,    VGPR v0..3 hold K = 0..7,  v4..7 hold K = 16..23
//   lanes 16-31: row = lane-16, VGPR v0..3 hold K = 8..15, v4..7 hold K = 24..31
// => per lane: two contiguous 8-element (16B) loads at k0+kh and k0+16+kh.
// ---------------------------------------------------------------------------
DEV const __bf16* frag_ptr(const __bf16* base, int ld, int r0, int k0, int lane) {
  int r  = lane & 15;
  int kh = (lane >> 4) * 8;
  return base + (size_t)(r0 + r) * (size_t)ld + (size_t)(k0 + kh);
}

DEV v16bf load_frag(const __bf16* p0) {
  v16bf f;
  ((v8bf*)&f)[0] = *(const v8bf*)(p0);
  ((v8bf*)&f)[1] = *(const v8bf*)(p0 + 16);
  return f;
}

DEV v8f wmma_bf16(v16bf a, v16bf b, v8f c) {
  return __builtin_amdgcn_wmma_f32_16x16x32_bf16(
      /*neg_a=*/false, a, /*neg_b=*/false, b,
      /*c_mod=*/(short)0, c, /*reuse_a=*/false, /*reuse_b=*/false);
}

// Software-pipelined K-loop: next fragments in flight while current WMMA issues
DEV v8f gemm_loop(const __bf16* pa, const __bf16* pb, int K) {
  v8f acc = {};
  v16bf a0 = load_frag(pa);
  v16bf b0 = load_frag(pb);
  for (int k0 = 32; k0 < K; k0 += 32) {
    v16bf a1 = load_frag(pa + k0);
    v16bf b1 = load_frag(pb + k0);
    acc = wmma_bf16(a0, b0, acc);
    a0 = a1; b0 = b1;
  }
  return wmma_bf16(a0, b0, acc);
}

// ---------------------------------------------------------------------------
// Weight f32 -> bf16 conversion
// ---------------------------------------------------------------------------
__global__ void k_tobf(const float* __restrict__ src, __bf16* __restrict__ dst, int n) {
  int i = blockIdx.x * 256 + threadIdx.x;
  if (i < n) dst[i] = (__bf16)src[i];
}

// ---------------------------------------------------------------------------
// Fused LayerNorm(features) + keys = feats@Wk^T + vT = (feats@Wv^T)^T
// one wave per 16-row tile of the [B*F, D] feature matrix
// ---------------------------------------------------------------------------
__global__ __launch_bounds__(32)
void k_lnproj(const float* __restrict__ feat,
              const float* __restrict__ g, const float* __restrict__ bta,
              const __bf16* __restrict__ Wk, const __bf16* __restrict__ Wv,
              __bf16* __restrict__ keys, __bf16* __restrict__ vT) {
  __shared__ __align__(16) __bf16 lds[16 * DD];
  int lane = threadIdx.x;
  size_t gr0 = (size_t)blockIdx.x * 16;

  // LayerNorm 16 rows into LDS as bf16
  for (int r = 0; r < 16; ++r) {
    const float* row = feat + (gr0 + r) * DD;
    float x[8]; float s = 0.f, s2 = 0.f;
#pragma unroll
    for (int i = 0; i < 8; ++i) { x[i] = row[lane * 8 + i]; s += x[i]; s2 += x[i] * x[i]; }
    s = wave_sum(s); s2 = wave_sum(s2);
    float mu  = s * (1.f / DD);
    float var = s2 * (1.f / DD) - mu * mu;
    float inv = rsqrtf(var + LNEPS);
#pragma unroll
    for (int i = 0; i < 8; ++i) {
      int c = lane * 8 + i;
      lds[r * DD + c] = (__bf16)((x[i] - mu) * inv * g[c] + bta[c]);
    }
  }
  __syncthreads();

  // preload the 8 A-fragments for this row tile (reused across 32 out tiles)
  v16bf afr[8];
  const __bf16* pa = frag_ptr(lds, DD, 0, 0, lane);
#pragma unroll
  for (int kk = 0; kk < 8; ++kk) afr[kk] = load_frag(pa + 32 * kk);

  int b  = (int)(gr0 >> 12);       // /4096
  int f0 = (int)(gr0 & 4095);
  int col_l = lane & 15, rhi = (lane >> 4) * 8;

  for (int ct = 0; ct < 32; ++ct) {
    bool isK = ct < 16;
    int n0 = (ct & 15) * 16;
    const __bf16* W = isK ? Wk : Wv;
    const __bf16* pb = frag_ptr(W, DD, n0, 0, lane);
    // double-buffered B fragments
    v16bf b0 = load_frag(pb);
    v8f acc = {};
#pragma unroll
    for (int kk = 0; kk < 7; ++kk) {
      v16bf b1 = load_frag(pb + 32 * (kk + 1));
      acc = wmma_bf16(afr[kk], b0, acc);
      b0 = b1;
    }
    acc = wmma_bf16(afr[7], b0, acc);

    int col = n0 + col_l;
    if (isK) {
#pragma unroll
      for (int r = 0; r < 8; ++r)
        keys[(gr0 + r + rhi) * DD + col] = (__bf16)acc[r];
    } else {
#pragma unroll
      for (int r = 0; r < 8; ++r)
        vT[((size_t)b * DD + col) * FF + (f0 + r + rhi)] = (__bf16)acc[r];
    }
  }
}

// ---------------------------------------------------------------------------
// Generic WMMA GEMM: C[m,n] = scale * sum_k A[m,k]*S[n,k] (+bias[n]) (+addend)
// one wave per 16x16 tile; grid = (Mtiles, Ntiles); ldc = Ntiles*16
// ---------------------------------------------------------------------------
__global__ __launch_bounds__(32)
void k_gemm(const __bf16* __restrict__ A, const __bf16* __restrict__ S,
            int K, int ldc,
            const float* __restrict__ bias, const float* __restrict__ addend,
            float scale, int relu,
            __bf16* __restrict__ outb, float* __restrict__ outf) {
  int lane = threadIdx.x;
  int m0 = blockIdx.x * 16, n0 = blockIdx.y * 16;
  v8f acc = gemm_loop(frag_ptr(A, K, m0, 0, lane),
                      frag_ptr(S, K, n0, 0, lane), K);
  int col = n0 + (lane & 15), rhi = (lane >> 4) * 8;
  float bv = bias ? bias[col] : 0.f;
#pragma unroll
  for (int r = 0; r < 8; ++r) {
    size_t off = (size_t)(m0 + r + rhi) * (size_t)ldc + col;
    float v = acc[r] * scale + bv;
    if (addend) v += addend[off];
    if (relu)   v = fmaxf(v, 0.f);
    if (outf) outf[off] = v;
    if (outb) outb[off] = (__bf16)v;
  }
}

// ---------------------------------------------------------------------------
// dots[b] = q[b] @ keys[b]^T  (M=16, N=4096, K=256)
// A-fragments preloaded once, reused across 4 N-tiles per wave. grid=(B, F/64)
// ---------------------------------------------------------------------------
__global__ __launch_bounds__(32)
void k_dots(const __bf16* __restrict__ q, const __bf16* __restrict__ keys,
            float* __restrict__ dots) {
  int lane = threadIdx.x;
  int b = blockIdx.x;
  int n0base = blockIdx.y * 64;
  const __bf16* Ab = q + (size_t)b * SP * DD;
  const __bf16* Sb = keys + (size_t)b * FF * DD;

  v16bf afr[8];
  const __bf16* pa = frag_ptr(Ab, DD, 0, 0, lane);
#pragma unroll
  for (int kk = 0; kk < 8; ++kk) afr[kk] = load_frag(pa + 32 * kk);

  int col_l = lane & 15, rhi = (lane >> 4) * 8;
  for (int t = 0; t < 4; ++t) {
    int n0 = n0base + t * 16;
    const __bf16* pb = frag_ptr(Sb, DD, n0, 0, lane);
    v16bf b0 = load_frag(pb);
    v8f acc = {};
#pragma unroll
    for (int kk = 0; kk < 7; ++kk) {
      v16bf b1 = load_frag(pb + 32 * (kk + 1));
      acc = wmma_bf16(afr[kk], b0, acc);
      b0 = b1;
    }
    acc = wmma_bf16(afr[7], b0, acc);
#pragma unroll
    for (int r = 0; r < 8; ++r)
      dots[(size_t)b * SP * FF + (size_t)(r + rhi) * FF + n0 + col_l] = acc[r];
  }
}

// ---------------------------------------------------------------------------
// Batched GEMM (M=16 per batch): grid = (B, Ntiles)
// updates: A=attn[b](16xF), S=vT[b] (DxF, ld=K=F) -> bf16 out
// ---------------------------------------------------------------------------
__global__ __launch_bounds__(32)
void k_gemm_bat(const __bf16* __restrict__ A, size_t sA,
                const __bf16* __restrict__ S, size_t sS,
                int K, int ldc, size_t sC, float scale,
                __bf16* __restrict__ outb, float* __restrict__ outf) {
  int lane = threadIdx.x;
  int b = blockIdx.x, n0 = blockIdx.y * 16;
  v8f acc = gemm_loop(frag_ptr(A + (size_t)b * sA, K, 0,  0, lane),
                      frag_ptr(S + (size_t)b * sS, K, n0, 0, lane), K);
  int col = n0 + (lane & 15), rhi = (lane >> 4) * 8;
#pragma unroll
  for (int r = 0; r < 8; ++r) {
    size_t off = (size_t)b * sC + (size_t)(r + rhi) * (size_t)ldc + col;
    float v = acc[r] * scale;
    if (outf) outf[off] = v;
    if (outb) outb[off] = (__bf16)v;
  }
}

// ---------------------------------------------------------------------------
// Slot LayerNorm (padded to 16 rows/batch); grid = B*16 rows, 256 threads
// ---------------------------------------------------------------------------
__global__ __launch_bounds__(256)
void k_ln_slots(const float* __restrict__ src, int sstride,
                const float* __restrict__ g, const float* __restrict__ bta,
                float* __restrict__ snf, __bf16* __restrict__ snb) {
  __shared__ float red[256];
  int b = blockIdx.x >> 4, s = blockIdx.x & 15, d = threadIdx.x;
  size_t off = (size_t)blockIdx.x * DD + d;
  if (s >= SS) { snf[off] = 0.f; snb[off] = (__bf16)0.f; return; }
  float x  = src[((size_t)b * sstride + s) * DD + d];
  float sm = block_sum(x, red);
  float s2 = block_sum(x * x, red);
  float mu  = sm * (1.f / DD);
  float var = s2 * (1.f / DD) - mu * mu;
  float inv = rsqrtf(var + LNEPS);
  float y = (x - mu) * inv * g[d] + bta[d];
  snf[off] = y;
  snb[off] = (__bf16)y;
}

// ---------------------------------------------------------------------------
// Softmax over the SLOT axis (S=11) per (b,f) column; writes pre_norm_attn
// ---------------------------------------------------------------------------
__global__ __launch_bounds__(256)
void k_softmax(const float* __restrict__ dots, float* __restrict__ pna) {
  int b = blockIdx.x;
  int f = blockIdx.y * 256 + threadIdx.x;
  float p[SS]; float m = -1e30f;
#pragma unroll
  for (int s = 0; s < SS; ++s) { p[s] = dots[((size_t)b * SP + s) * FF + f]; m = fmaxf(m, p[s]); }
  float sum = 0.f;
#pragma unroll
  for (int s = 0; s < SS; ++s) { p[s] = __expf(p[s] - m); sum += p[s]; }
  float r = 1.f / sum;
#pragma unroll
  for (int s = 0; s < SS; ++s) pna[((size_t)b * SS + s) * FF + f] = p[s] * r;
}

// Row sums over F per valid (b,s); grid = B*S rows
__global__ __launch_bounds__(256)
void k_rowsum(const float* __restrict__ pna, float* __restrict__ rowsum) {
  __shared__ float red[256];
  int row = blockIdx.x;                // b*11+s
  float s = 0.f;
  for (int i = threadIdx.x; i < FF; i += 256) s += pna[(size_t)row * FF + i];
  s = block_sum(s, red);
  if (threadIdx.x == 0) {
    int b = row / SS, ss = row % SS;
    rowsum[b * SP + ss] = s + (float)FF * EPSA;
  }
}

// attn_bf16[b][s_pad][f] = (p + EPS) / rowsum  (0 on pad rows)
__global__ __launch_bounds__(256)
void k_attn(const float* __restrict__ pna, const float* __restrict__ rowsum,
            __bf16* __restrict__ attn_bf) {
  int row = blockIdx.x; int b = row >> 4, s = row & 15;
  size_t f = (size_t)blockIdx.y * 256 + threadIdx.x;
  float v = 0.f;
  if (s < SS) v = (pna[((size_t)b * SS + s) * FF + f] + EPSA) / rowsum[row];
  attn_bf[(size_t)row * FF + f] = (__bf16)v;
}

// ---------------------------------------------------------------------------
// GRUCell (r,z,n order) + LayerNorm(new) -> bf16 MLP input; grid = B*16 rows
// ---------------------------------------------------------------------------
__global__ __launch_bounds__(256)
void k_gru(const float* __restrict__ gx, const float* __restrict__ gh,
           const float* __restrict__ bih, const float* __restrict__ bhh,
           const float* __restrict__ snf,
           const float* __restrict__ lg, const float* __restrict__ lb,
           float* __restrict__ newf, __bf16* __restrict__ hmlp) {
  __shared__ float red[256];
  int row = blockIdx.x; int s = row & 15; int d = threadIdx.x;
  size_t o256 = (size_t)row * DD + d;
  if (s >= SS) { newf[o256] = 0.f; hmlp[o256] = (__bf16)0.f; return; }
  size_t g0 = (size_t)row * (3 * DD);
  float r = sigmoidf(gx[g0 + d]         + bih[d]          + gh[g0 + d]         + bhh[d]);
  float z = sigmoidf(gx[g0 + DD + d]    + bih[DD + d]     + gh[g0 + DD + d]    + bhh[DD + d]);
  float n = tanhf  (gx[g0 + 2*DD + d]   + bih[2*DD + d]   + r * (gh[g0 + 2*DD + d] + bhh[2*DD + d]));
  float h = snf[o256];
  float nv = (1.f - z) * n + z * h;
  newf[o256] = nv;
  float sm = block_sum(nv, red);
  float s2 = block_sum(nv * nv, red);
  float mu  = sm * (1.f / DD);
  float var = s2 * (1.f / DD) - mu * mu;
  float inv = rsqrtf(var + LNEPS);
  hmlp[o256] = (__bf16)((nv - mu) * inv * lg[d] + lb[d]);
}

// de-pad final slots -> d_out[0 : B*S*D]
__global__ __launch_bounds__(256)
void k_copyout(const float* __restrict__ slots_cur, float* __restrict__ out) {
  int row = blockIdx.x;                // b*11+s
  int b = row / SS, s = row % SS, d = threadIdx.x;
  out[(size_t)row * DD + d] = slots_cur[((size_t)b * SP + s) * DD + d];
}

// ---------------------------------------------------------------------------
extern "C" void kernel_launch(void* const* d_in, const int* in_sizes, int n_in,
                              void* d_out, int out_size, void* d_ws, size_t ws_size,
                              hipStream_t stream) {
  const float* slots_in = (const float*)d_in[0];
  const float* features = (const float*)d_in[1];
  const float* Wk   = (const float*)d_in[2];
  const float* Wv   = (const float*)d_in[3];
  const float* Wq   = (const float*)d_in[4];
  const float* Wih  = (const float*)d_in[5];
  const float* Whh  = (const float*)d_in[6];
  const float* bih  = (const float*)d_in[7];
  const float* bhh  = (const float*)d_in[8];
  const float* lnfg = (const float*)d_in[9];
  const float* lnfb = (const float*)d_in[10];
  const float* lnsg = (const float*)d_in[11];
  const float* lnsb = (const float*)d_in[12];
  const float* lnmg = (const float*)d_in[13];
  const float* lnmb = (const float*)d_in[14];
  const float* W1   = (const float*)d_in[15];
  const float* b1   = (const float*)d_in[16];
  const float* W2   = (const float*)d_in[17];
  const float* b2   = (const float*)d_in[18];

  float* out_slots = (float*)d_out;
  float* out_attn  = (float*)d_out + BB * SS * DD;   // 90112 offset

  // workspace bump allocator (256B aligned)
  char* p = (char*)d_ws;
  auto alloc = [&](size_t bytes) -> void* {
    void* r = (void*)p;
    p += (bytes + 255) & ~(size_t)255;
    return r;
  };
  __bf16* keys_bf = (__bf16*)alloc((size_t)BB * FF * DD * 2);
  __bf16* vT_bf   = (__bf16*)alloc((size_t)BB * DD * FF * 2);
  __bf16* Wk_bf   = (__bf16*)alloc(DD * DD * 2);
  __bf16* Wv_bf   = (__bf16*)alloc(DD * DD * 2);
  __bf16* Wq_bf   = (__bf16*)alloc(DD * DD * 2);
  __bf16* Wih_bf  = (__bf16*)alloc(3 * DD * DD * 2);
  __bf16* Whh_bf  = (__bf16*)alloc(3 * DD * DD * 2);
  __bf16* W1_bf   = (__bf16*)alloc(HID * DD * 2);
  __bf16* W2_bf   = (__bf16*)alloc(DD * HID * 2);
  float*  snf     = (float*)alloc((size_t)BB * SP * DD * 4);
  __bf16* snb     = (__bf16*)alloc((size_t)BB * SP * DD * 2);
  __bf16* q_bf    = (__bf16*)alloc((size_t)BB * SP * DD * 2);
  float*  dots    = (float*)alloc((size_t)BB * SP * FF * 4);
  float*  rowsum  = (float*)alloc(BB * SP * 4);
  __bf16* attn_bf = (__bf16*)alloc((size_t)BB * SP * FF * 2);
  __bf16* upd_bf  = (__bf16*)alloc((size_t)BB * SP * DD * 2);
  float*  gx      = (float*)alloc((size_t)BB * SP * 3 * DD * 4);
  float*  gh      = (float*)alloc((size_t)BB * SP * 3 * DD * 4);
  float*  newf    = (float*)alloc((size_t)BB * SP * DD * 4);
  __bf16* hmlp_bf = (__bf16*)alloc((size_t)BB * SP * DD * 2);
  __bf16* h1_bf   = (__bf16*)alloc((size_t)BB * SP * HID * 2);
  float*  slotsc  = (float*)alloc((size_t)BB * SP * DD * 4);

  const float SCALE = 1.0f / 16.0f;   // 256^-0.5

  // weights -> bf16
  auto tobf = [&](const float* s, __bf16* d, int n) {
    k_tobf<<<(n + 255) / 256, 256, 0, stream>>>(s, d, n);
  };
  tobf(Wk, Wk_bf, DD * DD);   tobf(Wv, Wv_bf, DD * DD);   tobf(Wq, Wq_bf, DD * DD);
  tobf(Wih, Wih_bf, 3 * DD * DD);  tobf(Whh, Whh_bf, 3 * DD * DD);
  tobf(W1, W1_bf, HID * DD);  tobf(W2, W2_bf, DD * HID);

  // LN(features) + K/V projection (heaviest kernel; WMMA)
  k_lnproj<<<(BB * FF) / 16, 32, 0, stream>>>(features, lnfg, lnfb,
                                              Wk_bf, Wv_bf, keys_bf, vT_bf);

  for (int it = 0; it < 3; ++it) {
    const float* ssrc = (it == 0) ? slots_in : slotsc;
    int sstride = (it == 0) ? SS : SP;
    k_ln_slots<<<BB * SP, 256, 0, stream>>>(ssrc, sstride, lnsg, lnsb, snf, snb);

    // q = LN(slots) @ Wq^T, SCALE folded in, bf16 out
    k_gemm<<<dim3(BB * SP / 16, DD / 16), 32, 0, stream>>>(
        snb, Wq_bf, DD, DD, nullptr, nullptr, SCALE, 0, q_bf, nullptr);

    // dots[b] = q[b] @ keys[b]^T
    k_dots<<<dim3(BB, FF / 64), 32, 0, stream>>>(q_bf, keys_bf, dots);

    k_softmax<<<dim3(BB, FF / 256), 256, 0, stream>>>(dots, out_attn);
    k_rowsum<<<BB * SS, 256, 0, stream>>>(out_attn, rowsum);
    k_attn<<<dim3(BB * SP, FF / 256), 256, 0, stream>>>(out_attn, rowsum, attn_bf);

    // updates[b] = attn[b] @ values[b]  (via vT, K=4096, N=256)
    k_gemm_bat<<<dim3(BB, DD / 16), 32, 0, stream>>>(
        attn_bf, (size_t)SP * FF, vT_bf, (size_t)DD * FF,
        FF, DD, (size_t)SP * DD, 1.0f, upd_bf, nullptr);

    // GRU gate GEMMs (biases added in k_gru)
    k_gemm<<<dim3(BB * SP / 16, 3 * DD / 16), 32, 0, stream>>>(
        upd_bf, Wih_bf, DD, 3 * DD, nullptr, nullptr, 1.0f, 0, nullptr, gx);
    k_gemm<<<dim3(BB * SP / 16, 3 * DD / 16), 32, 0, stream>>>(
        snb, Whh_bf, DD, 3 * DD, nullptr, nullptr, 1.0f, 0, nullptr, gh);

    k_gru<<<BB * SP, 256, 0, stream>>>(gx, gh, bih, bhh, snf, lnmg, lnmb, newf, hmlp_bf);

    // MLP: h1 = relu(hmlp @ W1^T + b1); slots = new + h1 @ W2^T + b2
    k_gemm<<<dim3(BB * SP / 16, HID / 16), 32, 0, stream>>>(
        hmlp_bf, W1_bf, DD, HID, b1, nullptr, 1.0f, 1, h1_bf, nullptr);
    k_gemm<<<dim3(BB * SP / 16, DD / 16), 32, 0, stream>>>(
        h1_bf, W2_bf, HID, DD, b2, newf, 1.0f, 0, nullptr, slotsc);
  }

  k_copyout<<<BB * SS, 256, 0, stream>>>(slotsc, out_slots);
}
